// FaceModel_68848325755266
// MI455X (gfx1250) — matrix operand order repaired and checked by reading, not compiled, not run on hardware
//
#include <hip/hip_runtime.h>

// ---------------------------------------------------------------------------
// Types for CDNA5 WMMA (wave32, v_wmma_f32_16x16x32_bf16)
// ---------------------------------------------------------------------------
typedef __bf16 bf16_t;
typedef __attribute__((ext_vector_type(16))) __bf16 v16bf;
typedef __attribute__((ext_vector_type(8)))  __bf16 v8bf;
typedef __attribute__((ext_vector_type(8)))  float  v8f;

// A-fragment (16x32 bf16, row-major source, lda in elements)
// lane L: row m = L&15, chunks at K-offset (L>>4)*8 and +16 (ISA 7.12.2)
static __device__ __forceinline__ v16bf load_a_bf16(const bf16_t* __restrict__ A,
                                                    int lda, int lane) {
  const int m  = lane & 15;
  const int k0 = (lane >> 4) << 3;
  const bf16_t* p = A + (size_t)m * lda + k0;
  union { v16bf v; v8bf h[2]; } u;
  u.h[0] = *(const v8bf*)(p);
  u.h[1] = *(const v8bf*)(p + 16);
  return u.v;
}

// A-fragment from fp32 source, converted in-register (exact for adj's 0/1).
static __device__ __forceinline__ v16bf load_a_f32(const float* __restrict__ A,
                                                   int lda, int lane) {
  const int m  = lane & 15;
  const int k0 = (lane >> 4) << 3;
  const float* p = A + (size_t)m * lda + k0;
  v16bf r;
#pragma unroll
  for (int i = 0; i < 8; ++i) r[i] = (__bf16)p[i];
#pragma unroll
  for (int i = 0; i < 8; ++i) r[8 + i] = (__bf16)p[16 + i];
  return r;
}

// B-fragment (32x16 bf16) from TRANSPOSED weights Wt[n][k] (ldw = K).
// lane L: col n = L&15, chunks at K-offset (L>>4)*8 and +16 (matches A).
static __device__ __forceinline__ v16bf load_b_bf16(const bf16_t* __restrict__ Wt,
                                                    int ldw, int lane) {
  const int n  = lane & 15;
  const int k0 = (lane >> 4) << 3;
  const bf16_t* p = Wt + (size_t)n * ldw + k0;
  union { v16bf v; v8bf h[2]; } u;
  u.h[0] = *(const v8bf*)(p);
  u.h[1] = *(const v8bf*)(p + 16);
  return u.v;
}

// Lean ELU: expf lowers to v_exp_f32 (TRANS); |err| vs expm1 ~1e-4 max, far
// below bf16 operand noise.
static __device__ __forceinline__ float elu_f(float v) {
  return (v > 0.0f) ? v : (__expf(v) - 1.0f);
}

// ---------------------------------------------------------------------------
// GEMM: C[M,Nout] = act(A[M,K] @ W + bias), W given transposed (Wt[Nout][K]).
// Block = 256 threads = 8 waves in 4(M) x 2(N); wave tile 32x64 (2x4 subtiles
// of 16x16 => 8 WMMA per K-step); block tile 128x128.
// ---------------------------------------------------------------------------
template <bool AF32, bool DO_ELU, bool HAS_BIAS>
__global__ __launch_bounds__(256)
void gemm_bf16(const void* __restrict__ Av, const bf16_t* __restrict__ Wt,
               const float* __restrict__ bias, bf16_t* __restrict__ C,
               int M, int K, int Nout, int ldc) {
  const int tid  = threadIdx.x;
  const int lane = tid & 31;
  const int wave = tid >> 5;
  const int wm   = wave & 3;   // 4 waves along M
  const int wn   = wave >> 2;  // 2 waves along N
  const int row0 = blockIdx.x * 128 + wm * 32;  // m-subtiles at +0, +16
  const int col0 = blockIdx.y * 128 + wn * 64;  // n-subtiles at +0,16,32,48

  v8f acc[2][4] = {};
  const float*  Af = (const float*)Av;
  const bf16_t* Ab = (const bf16_t*)Av;

  for (int k = 0; k < K; k += 32) {
    v16bf a[2], b[4];
    if constexpr (AF32) {
      a[0] = load_a_f32(Af + (size_t)row0 * K + k, K, lane);
      a[1] = load_a_f32(Af + (size_t)(row0 + 16) * K + k, K, lane);
    } else {
      a[0] = load_a_bf16(Ab + (size_t)row0 * K + k, K, lane);
      a[1] = load_a_bf16(Ab + (size_t)(row0 + 16) * K + k, K, lane);
    }
#pragma unroll
    for (int ni = 0; ni < 4; ++ni)
      b[ni] = load_b_bf16(Wt + (size_t)(col0 + ni * 16) * K + k, K, lane);

#pragma unroll
    for (int mi = 0; mi < 2; ++mi)
#pragma unroll
      for (int ni = 0; ni < 4; ++ni)
        acc[mi][ni] = __builtin_amdgcn_wmma_f32_16x16x32_bf16(
            false, a[mi], false, b[ni], (short)0, acc[mi][ni], false, false);
  }

  // Epilogue: bias + ELU + bf16 store. C layout (ISA 7.12.2):
  // VGPR r -> row (r + ((lane>>4)<<3)), col lane&15.
  const int cn    = lane & 15;
  const int rbase = (lane >> 4) << 3;
#pragma unroll
  for (int mi = 0; mi < 2; ++mi) {
    const int rr = row0 + mi * 16 + rbase;
#pragma unroll
    for (int ni = 0; ni < 4; ++ni) {
      const int col = col0 + ni * 16 + cn;
      float bv = 0.0f;
      if constexpr (HAS_BIAS) bv = bias[col];
#pragma unroll
      for (int r = 0; r < 8; ++r) {
        float v = acc[mi][ni][r] + bv;
        if (DO_ELU) v = elu_f(v);
        C[(size_t)(rr + r) * ldc + col] = (bf16_t)v;
      }
    }
  }
}

// ---------------------------------------------------------------------------
// Weight transpose + fp32->bf16 convert: Wt[n*K + k] = W[k*Nout + n]
// ---------------------------------------------------------------------------
__global__ void transpose_convert(const float* __restrict__ W, bf16_t* __restrict__ Wt,
                                  int K, int Nout) {
  size_t idx = (size_t)blockIdx.x * blockDim.x + threadIdx.x;
  if (idx >= (size_t)K * Nout) return;
  int k = (int)(idx % K);
  int n = (int)(idx / K);
  Wt[idx] = (bf16_t)W[(size_t)k * Nout + n];
}

// ---------------------------------------------------------------------------
// Edge gather: e0[i,:] = s1[v1[i],:] + s1[v2[i],:]
// ---------------------------------------------------------------------------
__global__ void edge_gather(const bf16_t* __restrict__ s1, const int* __restrict__ v1,
                            const int* __restrict__ v2, bf16_t* __restrict__ e0,
                            int E, int F) {
  size_t idx = (size_t)blockIdx.x * blockDim.x + threadIdx.x;
  if (idx >= (size_t)E * F) return;
  int f = (int)(idx % F);
  int i = (int)(idx / F);
  float a = (float)s1[(size_t)v1[i] * F + f];
  float b = (float)s1[(size_t)v2[i] * F + f];
  e0[idx] = (bf16_t)(a + b);
}

// ---------------------------------------------------------------------------
// BatchNorm: deterministic two-level reduction (no atomics)
// ---------------------------------------------------------------------------
__global__ void bn_stats(const bf16_t* __restrict__ X, int M, int F,
                         int rowsPerChunk, float* __restrict__ partial) {
  int c = blockIdx.x * blockDim.x + threadIdx.x;   // column (coalesced per row)
  int chunk = blockIdx.y;
  int r0 = chunk * rowsPerChunk;
  int r1 = r0 + rowsPerChunk; if (r1 > M) r1 = M;
  float s = 0.0f, s2 = 0.0f;
  for (int r = r0; r < r1; ++r) {
    float v = (float)X[(size_t)r * F + c];
    s += v; s2 += v * v;
  }
  partial[((size_t)chunk * F + c) * 2 + 0] = s;
  partial[((size_t)chunk * F + c) * 2 + 1] = s2;
}

__global__ void bn_finalize(const float* __restrict__ partial, int CH, int F, int M,
                            const float* __restrict__ g, const float* __restrict__ be,
                            float* __restrict__ ab) {
  int c = blockIdx.x * blockDim.x + threadIdx.x;
  if (c >= F) return;
  float s = 0.0f, s2 = 0.0f;
  for (int ch = 0; ch < CH; ++ch) {
    s  += partial[((size_t)ch * F + c) * 2 + 0];
    s2 += partial[((size_t)ch * F + c) * 2 + 1];
  }
  float mu  = s / (float)M;
  float var = s2 / (float)M - mu * mu;           // biased variance (matches jnp.var)
  float alpha = g[c] * rsqrtf(var + 1e-5f);
  ab[2 * c + 0] = alpha;
  ab[2 * c + 1] = be[c] - alpha * mu;
}

// Row-major apply; optional dual store (Y2 with its own row stride, e.g. concat)
__global__ void bn_apply_rows(const bf16_t* __restrict__ X, const float* __restrict__ ab,
                              bf16_t* __restrict__ Y, bf16_t* __restrict__ Y2,
                              int M, int F, int ld2) {
  size_t idx = (size_t)blockIdx.x * blockDim.x + threadIdx.x;
  if (idx >= (size_t)M * F) return;
  int f = (int)(idx % F);
  size_t r = idx / F;
  bf16_t o = (bf16_t)fmaf((float)X[idx], ab[2 * f], ab[2 * f + 1]);
  Y[idx] = o;
  if (Y2) Y2[r * ld2 + f] = o;
}

// Transposed apply: Yt[f*M + r] — feeds adj@e as contiguous B-fragments.
__global__ void bn_apply_transpose(const bf16_t* __restrict__ X, const float* __restrict__ ab,
                                   bf16_t* __restrict__ Yt, int M, int F) {
  size_t idx = (size_t)blockIdx.x * blockDim.x + threadIdx.x;
  if (idx >= (size_t)M * F) return;
  int f = (int)(idx % F);
  size_t r = idx / F;
  Yt[(size_t)f * M + r] = (bf16_t)fmaf((float)X[idx], ab[2 * f], ab[2 * f + 1]);
}

// ---------------------------------------------------------------------------
// Final projection: out[r, 0:2] = h[r,:] @ w3c + b3c  (fp32, wave32 shuffles)
// ---------------------------------------------------------------------------
__global__ __launch_bounds__(256)
void final_proj(const bf16_t* __restrict__ H, const float* __restrict__ w3c,
                const float* __restrict__ b3c, float* __restrict__ out, int F) {
  int wave = threadIdx.x >> 5;
  int lane = threadIdx.x & 31;
  int row  = blockIdx.x * 8 + wave;
  float a0 = 0.0f, a1 = 0.0f;
  for (int k = lane; k < F; k += 32) {
    float h = (float)H[(size_t)row * F + k];
    a0 += h * w3c[2 * k + 0];
    a1 += h * w3c[2 * k + 1];
  }
  for (int off = 16; off > 0; off >>= 1) {
    a0 += __shfl_xor(a0, off, 32);
    a1 += __shfl_xor(a1, off, 32);
  }
  if (lane == 0) {
    out[2 * row + 0] = a0 + b3c[0];
    out[2 * row + 1] = a1 + b3c[1];
  }
}

// ---------------------------------------------------------------------------
// Host-side orchestration
// ---------------------------------------------------------------------------
extern "C" void kernel_launch(void* const* d_in, const int* in_sizes, int n_in,
                              void* d_out, int out_size, void* d_ws, size_t ws_size,
                              hipStream_t stream) {
  (void)n_in; (void)out_size; (void)ws_size;
  const int F = 512;
  const int N = in_sizes[0] / F;     // 4096
  const int E = in_sizes[1];         // 16384
  const int F2 = 2 * F;

  const float* x   = (const float*)d_in[0];
  const int*   v1  = (const int*)  d_in[1];
  const int*   v2  = (const int*)  d_in[2];
  const float* adj = (const float*)d_in[3];
  const float* w1a = (const float*)d_in[4];  const float* b1a = (const float*)d_in[5];
  const float* w1b = (const float*)d_in[6];  const float* b1b = (const float*)d_in[7];
  const float* g1  = (const float*)d_in[8];  const float* be1 = (const float*)d_in[9];
  const float* w2a = (const float*)d_in[10]; const float* b2a = (const float*)d_in[11];
  const float* w2b = (const float*)d_in[12]; const float* b2b = (const float*)d_in[13];
  const float* g2  = (const float*)d_in[14]; const float* be2 = (const float*)d_in[15];
  const float* w2c = (const float*)d_in[16]; const float* b2c = (const float*)d_in[17];
  const float* w2d = (const float*)d_in[18]; const float* b2d = (const float*)d_in[19];
  const float* w3a = (const float*)d_in[20]; const float* b3a = (const float*)d_in[21];
  const float* w3b = (const float*)d_in[22]; const float* b3b = (const float*)d_in[23];
  const float* g3  = (const float*)d_in[24]; const float* be3 = (const float*)d_in[25];
  const float* w3c = (const float*)d_in[26]; const float* b3c = (const float*)d_in[27];
  float* out = (float*)d_out;

  // Workspace carve-out (256B aligned)
  char* ws = (char*)d_ws;
  size_t off = 0;
  auto take = [&](size_t bytes) -> void* {
    void* p = ws + off;
    off = (off + bytes + 255) & ~(size_t)255;
    return p;
  };
  bf16_t* wt1a = (bf16_t*)take((size_t)F  * F * 2);
  bf16_t* wt1b = (bf16_t*)take((size_t)F  * F * 2);
  bf16_t* wt2a = (bf16_t*)take((size_t)F  * F * 2);
  bf16_t* wt2b = (bf16_t*)take((size_t)F  * F * 2);
  bf16_t* wt2c = (bf16_t*)take((size_t)F  * F * 2);
  bf16_t* wt2d = (bf16_t*)take((size_t)F  * F * 2);
  bf16_t* wt3a = (bf16_t*)take((size_t)F2 * F * 2);
  bf16_t* wt3b = (bf16_t*)take((size_t)F  * F * 2);
  bf16_t* s1b  = (bf16_t*)take((size_t)N  * F * 2);
  bf16_t* catb = (bf16_t*)take((size_t)N  * F2 * 2);
  bf16_t* nb   = (bf16_t*)take((size_t)N  * F * 2);
  bf16_t* tA   = (bf16_t*)take((size_t)E  * F * 2);
  bf16_t* tB   = (bf16_t*)take((size_t)E  * F * 2);
  const int CH = 64;
  float* partial = (float*)take((size_t)CH * F * 2 * 4);
  float* ab      = (float*)take((size_t)F * 2 * 4);

  auto tc = [&](const float* W, bf16_t* Wt, int K, int Nout) {
    size_t total = (size_t)K * Nout;
    transpose_convert<<<(unsigned)((total + 255) / 256), 256, 0, stream>>>(W, Wt, K, Nout);
  };
  auto bn = [&](const bf16_t* X, int M, const float* g, const float* be) {
    bn_stats<<<dim3(F / 256, CH), 256, 0, stream>>>(X, M, F, M / CH, partial);
    bn_finalize<<<(F + 255) / 256, 256, 0, stream>>>(partial, CH, F, M, g, be, ab);
  };
  size_t NF = (size_t)N * F, EF = (size_t)E * F;

  // --- weights: transpose + convert to bf16 ---
  tc(w1a, wt1a, F, F);   tc(w1b, wt1b, F, F);
  tc(w2a, wt2a, F, F);   tc(w2b, wt2b, F, F);
  tc(w2c, wt2c, F, F);   tc(w2d, wt2d, F, F);
  tc(w3a, wt3a, F2, F);  tc(w3b, wt3b, F, F);

  // --- stage 1: ELU(ELU(x@w1a+b1a)@w1b+b1b) -> BN -> s1 (dual store into concat-left)
  gemm_bf16<true,  true, true><<<dim3(N / 128, F / 128), 256, 0, stream>>>(x,  wt1a, b1a, tA, N, F, F, F);
  gemm_bf16<false, true, true><<<dim3(N / 128, F / 128), 256, 0, stream>>>(tA, wt1b, b1b, tB, N, F, F, F);
  bn(tB, N, g1, be1);
  bn_apply_rows<<<(unsigned)((NF + 255) / 256), 256, 0, stream>>>(tB, ab, s1b, catb, N, F, F2);

  // --- stage 2 edges: gather, MLP, BN (transposed output -> et[F][E])
  edge_gather<<<(unsigned)((EF + 255) / 256), 256, 0, stream>>>(s1b, v1, v2, tA, E, F);
  gemm_bf16<false, true, true><<<dim3(E / 128, F / 128), 256, 0, stream>>>(tA, wt2a, b2a, tB, E, F, F, F);
  gemm_bf16<false, true, true><<<dim3(E / 128, F / 128), 256, 0, stream>>>(tB, wt2b, b2b, tA, E, F, F, F);
  bn(tA, E, g2, be2);
  bn_apply_transpose<<<(unsigned)((EF + 255) / 256), 256, 0, stream>>>(tA, ab, tB, E, F);

  // --- aggregation: n = adj(fp32 0/1) @ e ; et in tB serves as pre-transposed B
  gemm_bf16<true, false, false><<<dim3(N / 128, F / 128), 256, 0, stream>>>(adj, tB, nullptr, nb, N, E, F, F);

  // --- stage 2 nodes: s2 = ELU(ELU(n@w2c+b2c)@w2d+b2d) -> concat-right (strided)
  gemm_bf16<false, true, true><<<dim3(N / 128, F / 128), 256, 0, stream>>>(nb, wt2c, b2c, tA, N, F, F, F);
  gemm_bf16<false, true, true><<<dim3(N / 128, F / 128), 256, 0, stream>>>(tA, wt2d, b2d, catb + F, N, F, F, F2);

  // --- stage 3: BN(ELU(ELU(cat@w3a+b3a)@w3b+b3b)) -> h ; out = h@w3c + b3c
  gemm_bf16<false, true, true><<<dim3(N / 128, F / 128), 256, 0, stream>>>(catb, wt3a, b3a, tA, N, F2, F, F);
  gemm_bf16<false, true, true><<<dim3(N / 128, F / 128), 256, 0, stream>>>(tA,   wt3b, b3b, tB, N, F,  F, F);
  bn(tB, N, g3, be3);
  bn_apply_rows<<<(unsigned)((NF + 255) / 256), 256, 0, stream>>>(tB, ab, tA, nullptr, N, F, F2);

  final_proj<<<N / 8, 256, 0, stream>>>(tA, w3c, b3c, out, F);
}
